// StableVQVAEModel_73890617361027
// MI455X (gfx1250) — compile-verified, matrix operand order checked by reading, not run on hardware
//
#include <hip/hip_runtime.h>
#include <math.h>

typedef __attribute__((ext_vector_type(16))) _Float16 v16h;
typedef __attribute__((ext_vector_type(8)))  _Float16 v8h;
typedef __attribute__((ext_vector_type(8)))  float    v8f;

// Padded NHWC geometry for the 3x3 WMMA convs: spatial 66x66 (1-px zero halo), 64 ch.
#define PADHW   66
#define PB_PIX  (PADHW * PADHW)          // 4356 pixels per batch
#define PB_STRIDE (PB_PIX * 64)          // 278784 halfs per batch
#define PAD_HALFS (16 * PB_STRIDE)       // 4,460,544 halfs per buffer
#define PAD_BYTES (PAD_HALFS * 2)        // 8,921,088 bytes (multiple of 16)

// Load one lane's 16-half WMMA fragment: elements 0..7 = p[0..7], 8..15 = p[16..23].
// Matches CDNA5 16-bit A/B layout (ISA 7.12.2) when p = row_base + kbase (+32*kstep).
__device__ __forceinline__ v16h load_frag16(const _Float16* p) {
  v8h lo = *reinterpret_cast<const v8h*>(p);
  v8h hi = *reinterpret_cast<const v8h*>(p + 16);
  return __builtin_shufflevector(lo, hi, 0,1,2,3,4,5,6,7,8,9,10,11,12,13,14,15);
}

// ---------------------------------------------------------------- prep kernels
__global__ void k_prep(const float* __restrict__ cb, _Float16* __restrict__ cbF16,
                       float* __restrict__ cnorm, unsigned* __restrict__ hist,
                       float* __restrict__ loss) {
  int k = threadIdx.x;  // 512 codes
  float s = 0.0f;
  for (int i = 0; i < 64; ++i) {
    float v = cb[k * 64 + i];
    s += v * v;
    cbF16[k * 64 + i] = (_Float16)v;
  }
  cnorm[k] = s;
  hist[k] = 0u;
  if (k == 0) *loss = 0.0f;
}

// Permute OIHW (64,64,3,3) weights to f16 (64, K'=576) with K' = r*64 + ci
// (tap-major, channel-minor) so each 32-wide WMMA k-step is one tap x 32 channels.
__global__ void k_wperm(const float* __restrict__ src, _Float16* __restrict__ dst) {
  int t = blockIdx.x * blockDim.x + threadIdx.x;  // 64*576 = 36864
  if (t >= 36864) return;
  int m = t / 576;
  int kp = t - m * 576;
  int r = kp >> 6;       // tap 0..8
  int ci = kp & 63;      // channel
  dst[t] = (_Float16)src[m * 576 + ci * 9 + r];
}

__global__ void k_zero4(float4* __restrict__ p, int n) {
  int i = blockIdx.x * blockDim.x + threadIdx.x;
  if (i < n) p[i] = make_float4(0.0f, 0.0f, 0.0f, 0.0f);
}

// ---------------------------------------------------------------- encoder convs
// conv1: x(16,3,256,256) clamped[0,1] -> (16,32,128,128), 4x4 s2 p1, ReLU
__global__ void k_conv1(const float* __restrict__ x, const float* __restrict__ w1,
                        const float* __restrict__ b1, float* __restrict__ out) {
  int o = blockIdx.x * blockDim.x + threadIdx.x;
  if (o >= 16 * 32 * 128 * 128) return;
  int ow = o & 127; int t = o >> 7; int oh = t & 127; t >>= 7; int co = t & 31; int b = t >> 5;
  float s = b1[co];
  for (int ci = 0; ci < 3; ++ci)
    for (int kh = 0; kh < 4; ++kh) {
      int ih = oh * 2 - 1 + kh;
      if ((unsigned)ih >= 256u) continue;
      for (int kw = 0; kw < 4; ++kw) {
        int iw = ow * 2 - 1 + kw;
        if ((unsigned)iw >= 256u) continue;
        float v = x[((b * 3 + ci) * 256 + ih) * 256 + iw];
        v = fminf(fmaxf(v, 0.0f), 1.0f);
        s += w1[((co * 3 + ci) * 4 + kh) * 4 + kw] * v;
      }
    }
  out[o] = fmaxf(s, 0.0f);
}

// conv2: (16,32,128,128) NCHW f32 -> padded NHWC f16 interior, 4x4 s2 p1, ReLU
__global__ void k_conv2(const float* __restrict__ in, const float* __restrict__ w2,
                        const float* __restrict__ b2, _Float16* __restrict__ outPad) {
  int o = blockIdx.x * blockDim.x + threadIdx.x;
  if (o >= 16 * 64 * 64 * 64) return;
  int ow = o & 63; int t = o >> 6; int oh = t & 63; t >>= 6; int co = t & 63; int b = t >> 6;
  float s = b2[co];
  for (int ci = 0; ci < 32; ++ci) {
    const float* inb = in + (size_t)((b * 32 + ci) * 128) * 128;
    for (int kh = 0; kh < 4; ++kh) {
      int ih = oh * 2 - 1 + kh;
      if ((unsigned)ih >= 128u) continue;
      for (int kw = 0; kw < 4; ++kw) {
        int iw = ow * 2 - 1 + kw;
        if ((unsigned)iw >= 128u) continue;
        s += w2[((co * 32 + ci) * 4 + kh) * 4 + kw] * inb[ih * 128 + iw];
      }
    }
  }
  outPad[((size_t)(b * PB_PIX + (oh + 1) * PADHW + (ow + 1)) << 6) + co] =
      (_Float16)fmaxf(s, 0.0f);
}

// ---------------------------------------------------------------- WMMA implicit-GEMM 3x3 conv
// 64->64 ch, 3x3, s1 p1, spatial 64x64, B=16. Input halo-padded NHWC f16, weights
// (64,576) with K' = tap*64 + ci. GEMM: M=Cout=64, K=576, N=65536 pixels.
// One wave = 16x16 tile, 18 fully-unrolled k-steps. All tap/channel deltas are
// compile-time constants folded into load immediate offsets: the hot loop is
// 4x global_load_b128 + 1x v_wmma per step, no address VALU, no masking.
__global__ void k_conv3x3_wmma(const _Float16* __restrict__ inPad,
                               const _Float16* __restrict__ wF16,
                               const float* __restrict__ bias,
                               float* __restrict__ outF32,          // NCHW
                               _Float16* __restrict__ outF16,       // NCHW optional copy
                               int relu) {
  const int lane = threadIdx.x & 31;
  const int wave = (blockIdx.x * blockDim.x + threadIdx.x) >> 5;  // 16384 waves exactly
  const int cotile = wave & 3;
  const int ptile  = wave >> 2;
  const int l15 = lane & 15;
  const int kb  = (lane & 16) ? 8 : 0;
  const int p = ptile * 16 + l15;          // pixel this lane's B-column covers
  const int b = p >> 12;
  const int h = (p >> 6) & 63;
  const int w = p & 63;
  const int m = cotile * 16 + l15;         // A row (cout)
  const _Float16* wrow = wF16 + (size_t)m * 576 + kb;
  const _Float16* inb  = inPad + (size_t)b * PB_STRIDE
                       + ((size_t)(h * PADHW + w) << 6) + kb;   // tap (0,0) = (h,w) in pad
  v8f acc = {};
#pragma unroll
  for (int step = 0; step < 18; ++step) {
    const int r   = step >> 1;           // tap 0..8 (compile-time after unroll)
    const int cib = (step & 1) << 5;     // channel base 0 / 32
    const int dh  = r / 3, dw = r % 3;   // compile-time tap deltas
    v16h a  = load_frag16(wrow + step * 32);
    v16h bf = load_frag16(inb + ((dh * PADHW + dw) << 6) + cib);
    acc = __builtin_amdgcn_wmma_f32_16x16x32_f16(false, a, false, bf, (short)0, acc,
                                                 false, false);
  }
  const int half8 = (lane & 16) ? 8 : 0;  // C layout: VGPR i -> row i (lanes 0-15) / 8+i
#pragma unroll
  for (int i = 0; i < 8; ++i) {
    int co = cotile * 16 + half8 + i;
    float val = acc[i] + bias[co];
    if (relu) val = fmaxf(val, 0.0f);
    size_t o = (((size_t)(b * 64 + co) * 64 + h) << 6) + w;
    outF32[o] = val;
    if (outF16) outF16[o] = (_Float16)val;
  }
}

// ---------------------------------------------------------------- VQ distance GEMM + fused argmin
// score[n,k] = ||c_k||^2 - 2*(f_n . c_k)  (||f_n||^2 constant per row -> dropped)
__global__ void k_vq_argmin(const _Float16* __restrict__ encF16,  // flat rows (65536,64)
                            const _Float16* __restrict__ cbF16,   // (512,64)
                            const float* __restrict__ cnorm,      // (512)
                            int* __restrict__ idx) {
  const int lane = threadIdx.x & 31;
  const int wave = (blockIdx.x * blockDim.x + threadIdx.x) >> 5;  // 4096 row tiles exactly
  const int l15 = lane & 15;
  const int kb  = (lane & 16) ? 8 : 0;
  const int row = wave * 16 + l15;
  const _Float16* arow = encF16 + (size_t)row * 64;
  v16h a0 = load_frag16(arow + kb);
  v16h a1 = load_frag16(arow + 32 + kb);
  float rmin[8]; int ridx[8];
#pragma unroll
  for (int i = 0; i < 8; ++i) { rmin[i] = 3.4e38f; ridx[i] = 0; }
  for (int ct = 0; ct < 32; ++ct) {
    int n = ct * 16 + l15;
    const _Float16* brow = cbF16 + (size_t)n * 64;
    v16h b0 = load_frag16(brow + kb);
    v16h b1 = load_frag16(brow + 32 + kb);
    v8f acc = {};
    acc = __builtin_amdgcn_wmma_f32_16x16x32_f16(false, a0, false, b0, (short)0, acc,
                                                 false, false);
    acc = __builtin_amdgcn_wmma_f32_16x16x32_f16(false, a1, false, b1, (short)0, acc,
                                                 false, false);
    float cn = cnorm[n];
#pragma unroll
    for (int i = 0; i < 8; ++i) {
      float s = cn - 2.0f * acc[i];
      if (s < rmin[i]) { rmin[i] = s; ridx[i] = n; }
    }
  }
#pragma unroll
  for (int off = 8; off >= 1; off >>= 1) {   // masks < 16 stay within each half
#pragma unroll
    for (int i = 0; i < 8; ++i) {
      float ov = __shfl_xor(rmin[i], off, 32);
      int   oi = __shfl_xor(ridx[i], off, 32);
      if (ov < rmin[i] || (ov == rmin[i] && oi < ridx[i])) { rmin[i] = ov; ridx[i] = oi; }
    }
  }
  if (l15 == 0) {
    int rb = wave * 16 + ((lane & 16) ? 8 : 0);
#pragma unroll
    for (int i = 0; i < 8; ++i) idx[rb + i] = ridx[i];
  }
}

// ---------------------------------------------------------------- quantize + loss + histogram
// q_st forward == codebook[idx]; write clipped q into padded NHWC f16 for dw1 conv.
__global__ void k_quant(const int* __restrict__ idx, const float* __restrict__ cb,
                        const float* __restrict__ enc, _Float16* __restrict__ qdPad,
                        unsigned* __restrict__ hist, float* __restrict__ loss) {
  int t = blockIdx.x * blockDim.x + threadIdx.x;  // 4,194,304 exactly (NCHW linear)
  int n = t >> 6, w = t & 63;
  int id = idx[n];
  float qv = cb[id * 64 + w];
  float d = qv - enc[t];
  int b = t >> 18, c = (t >> 12) & 63, h = (t >> 6) & 63;
  qdPad[((size_t)(b * PB_PIX + (h + 1) * PADHW + (w + 1)) << 6) + c] =
      (_Float16)fminf(fmaxf(qv, -10.0f), 10.0f);
  __shared__ float sm[256];
  sm[threadIdx.x] = d * d;
  __syncthreads();
  for (int s = 128; s > 0; s >>= 1) {
    if ((int)threadIdx.x < s) sm[threadIdx.x] += sm[threadIdx.x + s];
    __syncthreads();
  }
  if (threadIdx.x == 0) atomicAdd(loss, sm[0]);
  if (w == 0) atomicAdd(&hist[id], 1u);
}

// ---------------------------------------------------------------- decoder transposed convs
// dt1: (16,64,64,64) -> (16,32,128,128), k4 s2 p1, ReLU. weight (in=64,out=32,4,4)
__global__ void k_dt1(const float* __restrict__ g, const float* __restrict__ wt,
                      const float* __restrict__ bt, float* __restrict__ out) {
  int o = blockIdx.x * blockDim.x + threadIdx.x;
  if (o >= 16 * 32 * 128 * 128) return;
  int ow = o & 127; int t = o >> 7; int oh = t & 127; t >>= 7; int co = t & 31; int b = t >> 5;
  float s = bt[co];
  for (int kh = 0; kh < 4; ++kh) {
    int hh = oh + 1 - kh;
    if (hh & 1) continue;
    int ih = hh >> 1;
    if ((unsigned)ih >= 64u) continue;
    for (int kw = 0; kw < 4; ++kw) {
      int ww = ow + 1 - kw;
      if (ww & 1) continue;
      int iw = ww >> 1;
      if ((unsigned)iw >= 64u) continue;
      for (int ci = 0; ci < 64; ++ci)
        s += g[((b * 64 + ci) * 64 + ih) * 64 + iw] * wt[((ci * 32 + co) * 4 + kh) * 4 + kw];
    }
  }
  out[o] = fmaxf(s, 0.0f);
}

// dt2: (16,32,128,128) -> (16,3,256,256), k4 s2 p1, sigmoid+clip. weight (32,3,4,4)
__global__ void k_dt2(const float* __restrict__ g, const float* __restrict__ wt,
                      const float* __restrict__ bt, float* __restrict__ out) {
  int o = blockIdx.x * blockDim.x + threadIdx.x;
  if (o >= 16 * 3 * 256 * 256) return;
  int ow = o & 255; int t = o >> 8; int oh = t & 255; t >>= 8; int co = t % 3; int b = t / 3;
  float s = bt[co];
  for (int kh = 0; kh < 4; ++kh) {
    int hh = oh + 1 - kh;
    if (hh & 1) continue;
    int ih = hh >> 1;
    if ((unsigned)ih >= 128u) continue;
    for (int kw = 0; kw < 4; ++kw) {
      int ww = ow + 1 - kw;
      if (ww & 1) continue;
      int iw = ww >> 1;
      if ((unsigned)iw >= 128u) continue;
      for (int ci = 0; ci < 32; ++ci)
        s += g[((b * 32 + ci) * 128 + ih) * 128 + iw] * wt[((ci * 3 + co) * 4 + kh) * 4 + kw];
    }
  }
  float sg = 1.0f / (1.0f + expf(-s));
  out[o] = fminf(fmaxf(sg, 0.0f), 1.0f);
}

// ---------------------------------------------------------------- finalize scalars & idx
__global__ void k_finalize(const unsigned* __restrict__ hist, const float* __restrict__ loss,
                           float* __restrict__ out2) {
  __shared__ float sm[512];
  int t = threadIdx.x;
  float p = (float)hist[t] * (1.0f / 65536.0f);
  sm[t] = -p * logf(p + 1e-10f);
  __syncthreads();
  for (int s = 256; s > 0; s >>= 1) {
    if (t < s) sm[t] += sm[t + s];
    __syncthreads();
  }
  if (t == 0) {
    out2[0] = 1.25f * (*loss) * (1.0f / 4194304.0f);  // 0.25*commit + codebook (same fwd)
    out2[1] = expf(sm[0]);                            // perplexity
  }
}

__global__ void k_idx_to_float(const int* __restrict__ idx, float* __restrict__ dst) {
  int n = blockIdx.x * blockDim.x + threadIdx.x;
  if (n < 65536) dst[n] = (float)idx[n];
}

// ---------------------------------------------------------------- launcher
extern "C" void kernel_launch(void* const* d_in, const int* in_sizes, int n_in,
                              void* d_out, int out_size, void* d_ws, size_t ws_size,
                              hipStream_t stream) {
  (void)in_sizes; (void)n_in; (void)out_size; (void)ws_size;
  const float* x    = (const float*)d_in[0];
  const float* w1   = (const float*)d_in[1];
  const float* b1   = (const float*)d_in[2];
  const float* w2   = (const float*)d_in[3];
  const float* b2   = (const float*)d_in[4];
  const float* w3   = (const float*)d_in[5];
  const float* b3   = (const float*)d_in[6];
  const float* cb   = (const float*)d_in[7];
  const float* dw1  = (const float*)d_in[8];
  const float* db1  = (const float*)d_in[9];
  const float* dt1  = (const float*)d_in[10];
  const float* dbt1 = (const float*)d_in[11];
  const float* dt2  = (const float*)d_in[12];
  const float* dbt2 = (const float*)d_in[13];
  float* out = (float*)d_out;

  char* ws = (char*)d_ws;
  size_t off = 0;
  auto carve = [&](size_t bytes) -> char* {
    char* p = ws + off;
    off = (off + bytes + 255) & ~(size_t)255;
    return p;
  };
  float*    buf1   = (float*)   carve(33554432);  // (16,32,128,128) f32: conv1 out, later dt1 out
  _Float16* h2Pad  = (_Float16*)carve(PAD_BYTES); // conv2 out, padded NHWC f16
  float*    enc    = (float*)   carve(16777216);  // (16,64,64,64) NCHW f32
  _Float16* encF16 = (_Float16*)carve(8388608);   // enc NCHW f16 (= flat VQ rows)
  _Float16* qdPad  = (_Float16*)carve(PAD_BYTES); // clipped quantized, padded NHWC f16
  float*    g1     = (float*)   carve(16777216);  // dw1 out NCHW f32
  _Float16* cbF16  = (_Float16*)carve(65536);     // codebook f16
  float*    cnorm  = (float*)   carve(2048);      // ||c_k||^2
  _Float16* w3F16  = (_Float16*)carve(73728);     // conv3 weights f16 (64,576) K'=r*64+ci
  _Float16* dw1F16 = (_Float16*)carve(73728);     // dw1 weights f16, same permutation
  int*      idxbuf = (int*)     carve(262144);    // 65536 indices
  unsigned* hist   = (unsigned*)carve(2048);      // 512 counts
  float*    lossp  = (float*)   carve(256);       // scalar accumulator

  const int padVec4 = PAD_BYTES / 16;             // 557,568 float4 per padded buffer
  k_prep<<<1, 512, 0, stream>>>(cb, cbF16, cnorm, hist, lossp);
  k_wperm<<<144, 256, 0, stream>>>(w3, w3F16);
  k_wperm<<<144, 256, 0, stream>>>(dw1, dw1F16);
  k_zero4<<<(padVec4 + 255) / 256, 256, 0, stream>>>((float4*)h2Pad, padVec4);
  k_zero4<<<(padVec4 + 255) / 256, 256, 0, stream>>>((float4*)qdPad, padVec4);
  k_conv1<<<32768, 256, 0, stream>>>(x, w1, b1, buf1);
  k_conv2<<<16384, 256, 0, stream>>>(buf1, w2, b2, h2Pad);
  k_conv3x3_wmma<<<2048, 256, 0, stream>>>(h2Pad, w3F16, b3, enc, encF16, 0);
  k_vq_argmin<<<512, 256, 0, stream>>>(encF16, cbF16, cnorm, idxbuf);
  k_quant<<<16384, 256, 0, stream>>>(idxbuf, cb, enc, qdPad, hist, lossp);
  k_conv3x3_wmma<<<2048, 256, 0, stream>>>(qdPad, dw1F16, db1, g1, (_Float16*)nullptr, 1);
  k_dt1<<<32768, 256, 0, stream>>>(g1, dt1, dbt1, buf1);
  k_dt2<<<12288, 256, 0, stream>>>(buf1, dt2, dbt2, out);
  k_finalize<<<1, 512, 0, stream>>>(hist, lossp, out + 3145728);
  k_idx_to_float<<<256, 256, 0, stream>>>(idxbuf, out + 3145730);
}
